// MultiHeadAttention_11390253269217
// MI455X (gfx1250) — compile-verified, matrix operand order checked by reading, not run on hardware
//
#include <hip/hip_runtime.h>
#include <hip/hip_bf16.h>
#include <stdint.h>

#define B_ 4
#define S_ 2048
#define E_ 1024
#define H_ 16
#define D_ 64

typedef __attribute__((ext_vector_type(16))) __bf16 v16bf;
typedef __attribute__((ext_vector_type(8)))  float  v8f;
typedef __attribute__((ext_vector_type(4)))  unsigned int v4u;
typedef __attribute__((ext_vector_type(8)))  int v8i_t;
typedef __attribute__((ext_vector_type(4)))  int v4i_t;

// ---- helpers -------------------------------------------------------------

__device__ __forceinline__ unsigned short f2bf(float f) {
  unsigned int u = __float_as_uint(f);
  u += 0x7fffu + ((u >> 16) & 1u);          // round-to-nearest-even
  return (unsigned short)(u >> 16);
}

// Build one 16x32 bf16 WMMA fragment from LDS (row-major, ld in halves).
// ISA 7.12.2 layout: lane = M (or N for B^T), VGPR j holds packed pair at
// K = 16*(j/4) + 8*(lane/16) + 2*(j%4).
__device__ __forceinline__ v16bf load_frag(const unsigned short* p, int ld, int lane) {
  const unsigned short* row = p + (lane & 15) * ld + ((lane >> 4) << 3);
  union { v16bf v; unsigned int u[8]; } f;
#pragma unroll
  for (int j = 0; j < 8; ++j) {
    int kk = ((j >> 2) << 4) + ((j & 3) << 1);
    f.u[j] = *(const unsigned int*)(row + kk);
  }
  return f.v;
}

__device__ __forceinline__ v8f wmma_bf16(v16bf a, v16bf b, v8f c) {
  return __builtin_amdgcn_wmma_f32_16x16x32_bf16(false, a, false, b, (short)0, c,
                                                 false, false);
}

// Tensor Data Mover: 2D tile load, 64 cols of bf16 (128B rows), 64 rows,
// row stride = 64 elements, LDS pitch padded 32dw + 4dw pad = 72 halves.
__device__ __forceinline__ void tdm_load_tile64x64(unsigned lds_off,
                                                   unsigned long long gaddr,
                                                   int tensor_rows) {
  v4u g0;
  g0[0] = 1u;                                         // count=1, user mode
  g0[1] = lds_off;                                    // LDS byte address
  g0[2] = (unsigned)(gaddr & 0xffffffffu);            // global_addr[31:0]
  g0[3] = (unsigned)((gaddr >> 32) & 0x01ffffffu)     // global_addr[56:32]
        | 0x80000000u;                                // type = 2 (image)
  v8i_t g1;
  g1[0] = (1 << 16)          // data_size = 2 bytes
        | (1 << 20)          // pad_enable
        | (4 << 22)          // pad_interval: 32 DWORDs between pads
        | (3 << 25);         // pad_amount: 4 DWORDs (-> 72-half LDS pitch)
  g1[1] = (64 << 16);                                 // tensor_dim0 = 64
  g1[2] = (tensor_rows & 0xffff) << 16;               // tensor_dim1 lo
  g1[3] = ((tensor_rows >> 16) & 0xffff) | (64 << 16);// dim1 hi | tile_dim0=64
  g1[4] = 64;                                         // tile_dim1=64, tile_dim2=0
  g1[5] = 64;                                         // tensor_dim0_stride = 64
  g1[6] = 0;
  g1[7] = 0;
  v4i_t z4 = {0, 0, 0, 0};
#if defined(__clang_major__) && (__clang_major__ >= 23)
  v8i_t z8 = {0, 0, 0, 0, 0, 0, 0, 0};
  __builtin_amdgcn_tensor_load_to_lds(g0, g1, z4, z4, z8, 0);
#else
  __builtin_amdgcn_tensor_load_to_lds(g0, g1, z4, z4, 0);
#endif
}

// ---- kernel 1: fp32 -> bf16 weight conversion ----------------------------

__global__ void cvt_f32_bf16(const float* __restrict__ src,
                             unsigned short* __restrict__ dst, int n) {
  int i = blockIdx.x * blockDim.x + threadIdx.x;
  int stride = gridDim.x * blockDim.x;
  for (; i < n; i += stride) dst[i] = f2bf(src[i]);
}

// ---- kernel 2: per-head QKV projection -----------------------------------
// C[b,h,i,n] = sum_k x[b,i,k] * W[h,k,n] + bias[h,n], out bf16 [B,H,S,D]
// Block tile 128x64, 8 waves, 32x32 register tile per wave (4 WMMA / k-step).

__global__ __launch_bounds__(256) void qkv_gemm(
    const float* __restrict__ x,
    const unsigned short* __restrict__ wqp,
    const unsigned short* __restrict__ wkp,
    const unsigned short* __restrict__ wvp,
    const float* __restrict__ bqp, const float* __restrict__ bkp,
    const float* __restrict__ bvp,
    unsigned short* __restrict__ Qo, unsigned short* __restrict__ Ko,
    unsigned short* __restrict__ Vo) {
  __shared__ unsigned short Asm[128 * 40];  // 128 rows x 32 k (+pad)
  __shared__ unsigned short Bsm[64 * 40];   // 64 n    x 32 k (transposed W)

  const int i0 = blockIdx.x * 128;
  const int sel = blockIdx.y % 3;
  const int h = blockIdx.y / 3;
  const int b = blockIdx.z;

  const unsigned short* w = (sel == 0) ? wqp : (sel == 1) ? wkp : wvp;
  const float* bias = (sel == 0) ? bqp : (sel == 1) ? bkp : bvp;
  unsigned short* out = (sel == 0) ? Qo : (sel == 1) ? Ko : Vo;

  const int tid = threadIdx.x;
  const int lane = tid & 31;
  const int wave = tid >> 5;
  const int wR = wave >> 1;   // 0..3 : 32-row block
  const int wC = wave & 1;    // 0..1 : 32-col block

  v8f c00 = {}, c01 = {}, c10 = {}, c11 = {};

  const int arow = tid >> 1, akb = (tid & 1) * 16;  // A staging: 128x32
  const int bk = tid >> 3, bnb = (tid & 7) * 8;     // B staging: transpose 32x64

  for (int k0 = 0; k0 < E_; k0 += 32) {
    {
      const float* xp = x + ((size_t)b * S_ + i0 + arow) * E_ + k0 + akb;
      __builtin_prefetch(xp + 32, 0, 1);
      float4 f0 = ((const float4*)xp)[0];
      float4 f1 = ((const float4*)xp)[1];
      float4 f2 = ((const float4*)xp)[2];
      float4 f3 = ((const float4*)xp)[3];
      union { uint4 v[2]; unsigned short s[16]; } pk;
      pk.s[0]  = f2bf(f0.x); pk.s[1]  = f2bf(f0.y);
      pk.s[2]  = f2bf(f0.z); pk.s[3]  = f2bf(f0.w);
      pk.s[4]  = f2bf(f1.x); pk.s[5]  = f2bf(f1.y);
      pk.s[6]  = f2bf(f1.z); pk.s[7]  = f2bf(f1.w);
      pk.s[8]  = f2bf(f2.x); pk.s[9]  = f2bf(f2.y);
      pk.s[10] = f2bf(f2.z); pk.s[11] = f2bf(f2.w);
      pk.s[12] = f2bf(f3.x); pk.s[13] = f2bf(f3.y);
      pk.s[14] = f2bf(f3.z); pk.s[15] = f2bf(f3.w);
      *(uint4*)(&Asm[arow * 40 + akb])     = pk.v[0];
      *(uint4*)(&Asm[arow * 40 + akb + 8]) = pk.v[1];
    }
    {
      const unsigned short* wp = w + ((size_t)h * E_ + k0 + bk) * D_ + bnb;
      union { uint4 v; unsigned short s[8]; } pk;
      pk.v = *(const uint4*)wp;
#pragma unroll
      for (int j = 0; j < 8; ++j) Bsm[(bnb + j) * 40 + bk] = pk.s[j];
    }
    __syncthreads();
    v16bf a0 = load_frag(Asm + (wR * 32) * 40, 40, lane);
    v16bf a1 = load_frag(Asm + (wR * 32 + 16) * 40, 40, lane);
    v16bf b0 = load_frag(Bsm + (wC * 32) * 40, 40, lane);
    v16bf b1 = load_frag(Bsm + (wC * 32 + 16) * 40, 40, lane);
    c00 = wmma_bf16(a0, b0, c00);
    c01 = wmma_bf16(a0, b1, c01);
    c10 = wmma_bf16(a1, b0, c10);
    c11 = wmma_bf16(a1, b1, c11);
    __syncthreads();
  }

  const int n0 = wC * 32 + (lane & 15);
  const float bias0 = bias[h * D_ + n0];
  const float bias1 = bias[h * D_ + n0 + 16];
  unsigned short* op = out + (size_t)(b * H_ + h) * S_ * D_;
#pragma unroll
  for (int r = 0; r < 8; ++r) {
    int i = i0 + wR * 32 + r + ((lane >> 4) << 3);
    op[(size_t)i * D_ + n0]             = f2bf(c00[r] + bias0);
    op[(size_t)i * D_ + n0 + 16]        = f2bf(c01[r] + bias1);
    op[(size_t)(i + 16) * D_ + n0]      = f2bf(c10[r] + bias0);
    op[(size_t)(i + 16) * D_ + n0 + 16] = f2bf(c11[r] + bias1);
  }
}

// ---- kernel 3: flash-style causal attention per (b,h,64-row tile) --------
// Q and K tiles are staged into LDS by the Tensor Data Mover (wave 0 issues,
// TENSORcnt waited, barrier publishes). V needs a transpose -> manual path.

__global__ __launch_bounds__(256) void attn(
    const unsigned short* __restrict__ Q, const unsigned short* __restrict__ K,
    const unsigned short* __restrict__ V, unsigned short* __restrict__ Hout) {
  __shared__ unsigned short Qs[64 * 72];
  __shared__ unsigned short Ks[64 * 72];
  __shared__ unsigned short Vt[64 * 72];   // transposed: [d][key]
  __shared__ unsigned short Ps[64 * 72];   // probabilities bf16
  __shared__ float Ss[64 * 65];            // raw scores
  __shared__ float mrow[64], lrow[64], arow[64];

  const int i0 = blockIdx.x * 64;
  const int h = blockIdx.y;
  const int b = blockIdx.z;
  const int tid = threadIdx.x;
  const int lane = tid & 31;
  const int wave = tid >> 5;
  const int wR = wave >> 1;   // 16-row block of queries / output rows
  const int wC = wave & 1;    // 32-wide block of keys (scores) or d (output)

  const size_t headBase = (size_t)(b * H_ + h) * S_ * D_;
  const int srow = tid >> 2, sdb = (tid & 3) * 16;

  const unsigned QsOff = (unsigned)(unsigned long long)(uintptr_t)Qs;
  const unsigned KsOff = (unsigned)(unsigned long long)(uintptr_t)Ks;

  if (wave == 0) {  // TDM: stage Q tile once
    tdm_load_tile64x64(QsOff,
                       (unsigned long long)(uintptr_t)(Q + headBase + (size_t)i0 * D_),
                       S_ - i0);
    __builtin_amdgcn_s_wait_tensorcnt(0);
  }
  if (tid < 64) { mrow[tid] = -3.0e38f; lrow[tid] = 0.0f; }

  v8f o0 = {}; v8f o1 = {};
  __syncthreads();

  for (int j0 = 0; j0 <= i0; j0 += 64) {
    if (wave == 0) {  // TDM: stage K tile row-major (padded to 72-half pitch)
      tdm_load_tile64x64(KsOff,
                         (unsigned long long)(uintptr_t)(K + headBase + (size_t)j0 * D_),
                         S_ - j0);
    }
    {  // stage V transposed (manual; TDM has no transpose)
      const uint4* vp = (const uint4*)(V + headBase + (size_t)(j0 + srow) * D_ + sdb);
      union { uint4 v; unsigned short s[8]; } u0, u1;
      u0.v = vp[0]; u1.v = vp[1];
#pragma unroll
      for (int j = 0; j < 8; ++j) {
        Vt[(sdb + j) * 72 + srow]     = u0.s[j];
        Vt[(sdb + 8 + j) * 72 + srow] = u1.s[j];
      }
    }
    if (wave == 0) __builtin_amdgcn_s_wait_tensorcnt(0);
    __syncthreads();

    {  // S = Q K^T * 1/sqrt(D)
      v8f s0 = {}; v8f s1 = {};
#pragma unroll
      for (int dk = 0; dk < 64; dk += 32) {
        v16bf a  = load_frag(Qs + wR * 16 * 72 + dk, 72, lane);
        v16bf b0 = load_frag(Ks + (wC * 32) * 72 + dk, 72, lane);
        v16bf b1 = load_frag(Ks + (wC * 32 + 16) * 72 + dk, 72, lane);
        s0 = wmma_bf16(a, b0, s0);
        s1 = wmma_bf16(a, b1, s1);
      }
      const int n = lane & 15;
#pragma unroll
      for (int r = 0; r < 8; ++r) {
        int m = wR * 16 + r + ((lane >> 4) << 3);
        Ss[m * 65 + wC * 32 + n]      = s0[r] * 0.125f;
        Ss[m * 65 + wC * 32 + n + 16] = s1[r] * 0.125f;
      }
    }
    __syncthreads();

    if (tid < 64) {  // online softmax update per query row
      const int row = tid;
      const int gi = i0 + row;
      float mold = mrow[row];
      float mnew = mold;
#pragma unroll 4
      for (int j = 0; j < 64; ++j) {
        float s = (j0 + j <= gi) ? Ss[row * 65 + j] : -3.0e38f;
        mnew = fmaxf(mnew, s);
      }
      float alpha = __expf(mold - mnew);
      float lsum = 0.0f;
#pragma unroll 4
      for (int j = 0; j < 64; ++j) {
        float s = (j0 + j <= gi) ? Ss[row * 65 + j] : -3.0e38f;
        float p = __expf(s - mnew);
        lsum += p;
        Ps[row * 72 + j] = f2bf(p);
      }
      mrow[row] = mnew;
      lrow[row] = lrow[row] * alpha + lsum;
      arow[row] = alpha;
    }
    __syncthreads();

    {  // O = O*alpha + P V
#pragma unroll
      for (int r = 0; r < 8; ++r) {
        float al = arow[wR * 16 + r + ((lane >> 4) << 3)];
        o0[r] *= al;
        o1[r] *= al;
      }
#pragma unroll
      for (int kk = 0; kk < 64; kk += 32) {
        v16bf a  = load_frag(Ps + wR * 16 * 72 + kk, 72, lane);
        v16bf b0 = load_frag(Vt + (wC * 32) * 72 + kk, 72, lane);
        v16bf b1 = load_frag(Vt + (wC * 32 + 16) * 72 + kk, 72, lane);
        o0 = wmma_bf16(a, b0, o0);
        o1 = wmma_bf16(a, b1, o1);
      }
    }
    __syncthreads();
  }

  {  // normalize and write concat-head layout [B,S,E]
    const int n = lane & 15;
#pragma unroll
    for (int r = 0; r < 8; ++r) {
      int m = wR * 16 + r + ((lane >> 4) << 3);
      float inv = 1.0f / lrow[m];
      size_t base = ((size_t)b * S_ + i0 + m) * E_ + h * D_ + wC * 32 + n;
      Hout[base]      = f2bf(o0[r] * inv);
      Hout[base + 16] = f2bf(o1[r] * inv);
    }
  }
}

// ---- kernel 4: output projection: out = Hout * Wo^T + bo (fp32 out) ------
// Block tile 128x64, 32x32 per-wave register tile.

__global__ __launch_bounds__(256) void out_proj(
    const unsigned short* __restrict__ A,    // [B*S, E] bf16
    const unsigned short* __restrict__ Wo,   // [E, E]  bf16, row e holds k's
    const float* __restrict__ bo, float* __restrict__ out) {
  __shared__ unsigned short Asm[128 * 40];
  __shared__ unsigned short Bsm[64 * 40];    // Bsm[n][k] = Wo[n0+n][k0+k]

  const int n0 = blockIdx.x * 64;
  const int m0 = blockIdx.y * 128;
  const int tid = threadIdx.x;
  const int lane = tid & 31;
  const int wave = tid >> 5;
  const int wR = wave >> 1;
  const int wC = wave & 1;

  v8f c00 = {}, c01 = {}, c10 = {}, c11 = {};
  const int arow = tid >> 1, akb = (tid & 1) * 16;
  const int brow = tid >> 2, bkb = (tid & 3) * 8;

  for (int k0 = 0; k0 < E_; k0 += 32) {
    const unsigned short* ap = A + (size_t)(m0 + arow) * E_ + k0 + akb;
    const unsigned short* bp = Wo + (size_t)(n0 + brow) * E_ + k0 + bkb;
    __builtin_prefetch(ap + 32, 0, 1);
    *(uint4*)(&Asm[arow * 40 + akb])     = ((const uint4*)ap)[0];
    *(uint4*)(&Asm[arow * 40 + akb + 8]) = ((const uint4*)ap)[1];
    *(uint4*)(&Bsm[brow * 40 + bkb])     = *(const uint4*)bp;
    __syncthreads();
    v16bf a0 = load_frag(Asm + (wR * 32) * 40, 40, lane);
    v16bf a1 = load_frag(Asm + (wR * 32 + 16) * 40, 40, lane);
    v16bf b0 = load_frag(Bsm + (wC * 32) * 40, 40, lane);
    v16bf b1 = load_frag(Bsm + (wC * 32 + 16) * 40, 40, lane);
    c00 = wmma_bf16(a0, b0, c00);
    c01 = wmma_bf16(a0, b1, c01);
    c10 = wmma_bf16(a1, b0, c10);
    c11 = wmma_bf16(a1, b1, c11);
    __syncthreads();
  }
  const int nn = n0 + wC * 32 + (lane & 15);
  const float bias0 = bo[nn], bias1 = bo[nn + 16];
#pragma unroll
  for (int r = 0; r < 8; ++r) {
    int m = m0 + wR * 32 + r + ((lane >> 4) << 3);
    out[(size_t)m * E_ + nn]             = c00[r] + bias0;
    out[(size_t)m * E_ + nn + 16]        = c01[r] + bias1;
    out[(size_t)(m + 16) * E_ + nn]      = c10[r] + bias0;
    out[(size_t)(m + 16) * E_ + nn + 16] = c11[r] + bias1;
  }
}

// ---- host launcher -------------------------------------------------------

extern "C" void kernel_launch(void* const* d_in, const int* in_sizes, int n_in,
                              void* d_out, int out_size, void* d_ws, size_t ws_size,
                              hipStream_t stream) {
  (void)in_sizes; (void)n_in; (void)out_size; (void)ws_size;
  const float* x  = (const float*)d_in[0];
  const float* Wq = (const float*)d_in[1];
  const float* bq = (const float*)d_in[2];
  const float* Wk = (const float*)d_in[3];
  const float* bk = (const float*)d_in[4];
  const float* Wv = (const float*)d_in[5];
  const float* bv = (const float*)d_in[6];
  const float* Wo = (const float*)d_in[7];
  const float* bo = (const float*)d_in[8];
  float* out = (float*)d_out;

  char* ws = (char*)d_ws;
  size_t off = 0;
  const size_t wsz = (size_t)H_ * E_ * D_;      // per-QKV weight elems
  const size_t qsz = (size_t)B_ * H_ * S_ * D_; // per Q/K/V elems
  unsigned short* wq_bf = (unsigned short*)(ws + off); off += wsz * 2;
  unsigned short* wk_bf = (unsigned short*)(ws + off); off += wsz * 2;
  unsigned short* wv_bf = (unsigned short*)(ws + off); off += wsz * 2;
  unsigned short* wo_bf = (unsigned short*)(ws + off); off += (size_t)E_ * E_ * 2;
  unsigned short* Qb    = (unsigned short*)(ws + off); off += qsz * 2;
  unsigned short* Kb    = (unsigned short*)(ws + off); off += qsz * 2;
  unsigned short* Vb    = (unsigned short*)(ws + off); off += qsz * 2;
  unsigned short* Hb    = (unsigned short*)(ws + off); off += (size_t)B_ * S_ * E_ * 2;

  // 1) weight conversion
  cvt_f32_bf16<<<1024, 256, 0, stream>>>(Wq, wq_bf, (int)wsz);
  cvt_f32_bf16<<<1024, 256, 0, stream>>>(Wk, wk_bf, (int)wsz);
  cvt_f32_bf16<<<1024, 256, 0, stream>>>(Wv, wv_bf, (int)wsz);
  cvt_f32_bf16<<<1024, 256, 0, stream>>>(Wo, wo_bf, E_ * E_);

  // 2) QKV projections (block tile 128 rows x 64 cols)
  qkv_gemm<<<dim3(S_ / 128, 3 * H_, B_), 256, 0, stream>>>(
      x, wq_bf, wk_bf, wv_bf, bq, bk, bv, Qb, Kb, Vb);

  // 3) causal attention (TDM-staged Q/K tiles)
  attn<<<dim3(S_ / 64, H_, B_), 256, 0, stream>>>(Qb, Kb, Vb, Hb);

  // 4) output projection (block tile 128 x 64)
  out_proj<<<dim3(E_ / 64, (B_ * S_) / 128), 256, 0, stream>>>(Hb, wo_bf, bo, out);
}
